// FullASAAttention_76227079569866
// MI455X (gfx1250) — matrix operand order repaired and checked by reading, not compiled, not run on hardware
//
#include <hip/hip_runtime.h>
#include <hip/hip_bf16.h>
#include <math.h>

typedef __attribute__((ext_vector_type(16))) __bf16 v16bf;
typedef __attribute__((ext_vector_type(8)))  __bf16 v8bf;
typedef __attribute__((ext_vector_type(8)))  float  v8f;
typedef __attribute__((ext_vector_type(4)))  unsigned int v4u;
typedef __attribute__((ext_vector_type(4)))  int v4i;
typedef __attribute__((ext_vector_type(8)))  int v8i;

constexpr int kB = 4;
constexpr int kN = 2048;
constexpr int kD = 1024;
constexpr int kS = 3;
constexpr int kC = 32;
constexpr int kRows = kB * kN;   // 8192

// LDS A-tile layout written by TDM with padding: every 256 dwords (512 bf16) of
// data, 4 dwords of pad are inserted -> row stride = 512+8 dwords = 2080 bytes,
// and the second half of each K row is shifted by +16 bytes. 16B alignment kept.
constexpr int kLdsRowBytes = 2080;
constexpr int kLdsABytes   = 32 * kLdsRowBytes + 32;

__device__ __forceinline__ float softplusf(float x){
  return x > 20.f ? x : log1pf(__expf(x));
}

// ---- WMMA fragment helpers (layouts per CDNA5 ISA 7.12.2) ----

// B: 32x16 bf16 from K-major ("transposed") weights Bt[col][k], leading dim ldb.
__device__ __forceinline__ v16bf load_b_frag(const __bf16* __restrict__ Bt, int col0, int ldb,
                                             int k0, int lane){
  int n  = lane & 15;
  int kb = (lane >> 4) * 8;
  const __bf16* p = Bt + (size_t)(col0 + n) * ldb + (k0 + kb);
  v8bf lo = *reinterpret_cast<const v8bf*>(p);
  v8bf hi = *reinterpret_cast<const v8bf*>(p + 16);
  v16bf b;
#pragma unroll
  for (int i = 0; i < 8; ++i){ b[i] = lo[i]; b[8 + i] = hi[i]; }
  return b;
}

// A: 16x32 bf16 fragment from the TDM-staged (padded) LDS tile.
__device__ __forceinline__ v16bf lds_a_frag(const char* base, int rl0, int k0, int lane){
  int m = rl0 + (lane & 15);
  int k = k0 + ((lane >> 4) * 8);
  const char* p = base + m * kLdsRowBytes + k * 2 + ((k0 >= 512) ? 16 : 0);
  v8bf lo = *reinterpret_cast<const v8bf*>(p);
  v8bf hi = *reinterpret_cast<const v8bf*>(p + 32);   // k+16 elements
  v16bf a;
#pragma unroll
  for (int i = 0; i < 8; ++i){ a[i] = lo[i]; a[8 + i] = hi[i]; }
  return a;
}

#define WMMA_BF16(a, b, c) \
  __builtin_amdgcn_wmma_f32_16x16x32_bf16(false, (a), false, (b), (short)0, (c), false, false)

// ---- Tensor Data Mover: stage a 32 x kD bf16 tile (row-major, ld=kD) into LDS ----
// D# per CDNA5 ISA: group0 = {count/type/addrs},
// group1 = {data_size, pad, tensor dims/strides, tile dims}. Groups 2/3 zero (2-D).
__device__ __forceinline__ void tdm_load_rows(const __bf16* g, unsigned ldsoff,
                                              int row0, int nrows, int rows_total){
  unsigned long long ga = (unsigned long long)(g + (size_t)row0 * kD);
  v4u g0;
  g0[0] = 1u;                                            // count=1 (valid user desc)
  g0[1] = ldsoff;                                        // lds_addr (bytes)
  g0[2] = (unsigned)ga;                                  // global_addr[31:0]
  g0[3] = ((unsigned)(ga >> 32) & 0x01FFFFFFu) | 0x80000000u; // addr[56:32] | type=2
  v8i g1;
  g1[0] = (1 << 16)      // data_size = 1 -> 2-byte elements
        | (1 << 20)      // pad_enable
        | (7 << 22)      // pad_interval code 7 = 256 dwords
        | (3 << 25);     // pad_amount  code 3 = 4 dwords
  g1[1] = (kD & 0xFFFF) << 16;                               // tensor_dim0 lo16
  g1[2] = (kD >> 16) | ((rows_total & 0xFFFF) << 16);        // dim0 hi | dim1 lo
  g1[3] = (rows_total >> 16) | ((kD & 0xFFFF) << 16);        // dim1 hi | tile_dim0
  g1[4] = nrows;                                             // tile_dim1 (tile_dim2=0)
  g1[5] = kD;                                                // tensor_dim0_stride lo32
  g1[6] = 0;                                                 // stride hi | dim1_stride lo
  g1[7] = 0;
  v4i z4 = {0, 0, 0, 0};
#if defined(__clang_major__) && (__clang_major__ >= 23)
  v8i z8 = {0, 0, 0, 0, 0, 0, 0, 0};
  __builtin_amdgcn_tensor_load_to_lds(g0, g1, z4, z4, z8, 0);
#else
  __builtin_amdgcn_tensor_load_to_lds(g0, g1, z4, z4, 0);
#endif
}

// ----------------------------- prep kernels -----------------------------

__global__ void k_cvt_bf16(const float* __restrict__ in, __bf16* __restrict__ out, int n){
  int i = blockIdx.x * 256 + threadIdx.x;
  if (i < n) out[i] = (__bf16)in[i];
}

// Wt[n*K + k] = W[k*Ncol + n] (fp32 -> bf16, K-major for B fragments)
__global__ void k_transpose_bf16(const float* __restrict__ W, __bf16* __restrict__ Wt,
                                 int K, int Ncol){
  int i = blockIdx.x * 256 + threadIdx.x;
  if (i < K * Ncol){
    int n = i / K, k = i % K;
    Wt[(size_t)n * K + k] = (__bf16)W[(size_t)k * Ncol + n];
  }
}

// xbar[b][d] = mean_n x[b][n][d]
__global__ void k_colmean(const float* __restrict__ x, float* __restrict__ xbar){
  int i = blockIdx.x * 256 + threadIdx.x;
  if (i >= kB * kD) return;
  int b = i / kD, d = i % kD;
  const float* p = x + (size_t)b * kN * kD + d;
  float s = 0.f;
  for (int n = 0; n < kN; ++n) s += p[(size_t)n * kD];
  xbar[i] = s * (1.f / (float)kN);
}

// ctx[b][j] = xbar[b]·W_ctx[:,j] + b_ctx[j]   (mean pulled through the linear layer)
__global__ void k_ctx(const float* __restrict__ xbar, const float* __restrict__ Wc,
                      const float* __restrict__ bc, float* __restrict__ ctx){
  int i = blockIdx.x * 256 + threadIdx.x;
  if (i >= kB * kD) return;
  int b = i / kD, j = i % kD;
  const float* xr = xbar + b * kD;
  float s = bc[j];
  for (int d = 0; d < kD; ++d) s += xr[d] * Wc[(size_t)d * kD + j];
  ctx[i] = s;
}

// selctx[b][s] = ctx[b]·W_sel[D:,s] + b_sel[s]
__global__ void k_selctx(const float* __restrict__ ctx, const float* __restrict__ Wsel,
                         const float* __restrict__ bsel, float* __restrict__ selctx){
  int b = blockIdx.x / kS, s = blockIdx.x % kS;
  __shared__ float red[256];
  float p = 0.f;
  for (int j = threadIdx.x; j < kD; j += 256)
    p += ctx[b * kD + j] * Wsel[(size_t)(kD + j) * kS + s];
  red[threadIdx.x] = p; __syncthreads();
  for (int off = 128; off > 0; off >>= 1){
    if (threadIdx.x < off) red[threadIdx.x] += red[threadIdx.x + off];
    __syncthreads();
  }
  if (threadIdx.x == 0) selctx[b * kS + s] = red[0] + bsel[s];
}

// sense_w[row][s] = softmax_s( x[row]·W_sel[:D,s] + selctx[b][s] )  (one wave per row)
__global__ void k_sensew(const float* __restrict__ x, const float* __restrict__ Wsel,
                         const float* __restrict__ selctx, float* __restrict__ sw){
  int wave = threadIdx.x >> 5, lane = threadIdx.x & 31;
  int row = blockIdx.x * 8 + wave;
  int b = row / kN;
  const float* xr = x + (size_t)row * kD;
  float a0 = 0.f, a1 = 0.f, a2 = 0.f;
  for (int k = lane; k < kD; k += 32){
    float xv = xr[k];
    const float* w = Wsel + (size_t)k * kS;
    a0 += xv * w[0]; a1 += xv * w[1]; a2 += xv * w[2];
  }
#pragma unroll
  for (int off = 16; off > 0; off >>= 1){
    a0 += __shfl_down(a0, off, 32);
    a1 += __shfl_down(a1, off, 32);
    a2 += __shfl_down(a2, off, 32);
  }
  if (lane == 0){
    a0 += selctx[b * kS + 0]; a1 += selctx[b * kS + 1]; a2 += selctx[b * kS + 2];
    float m = fmaxf(a0, fmaxf(a1, a2));
    float e0 = __expf(a0 - m), e1 = __expf(a1 - m), e2 = __expf(a2 - m);
    float inv = 1.f / (e0 + e1 + e2);
    sw[row * 3 + 0] = e0 * inv; sw[row * 3 + 1] = e1 * inv; sw[row * 3 + 2] = e2 * inv;
  }
}

// ---- fused sense GEMM + mix. Block = 128 (4 waves) covers 32 rows x 64 cols.
// A tile (32 x 1024 bf16) staged to LDS once via TDM; 3 senses x 2 row-frags = 6 accs.
__global__ void __launch_bounds__(128) k_sense_gemm(
    const __bf16* __restrict__ xb, const __bf16* __restrict__ Wt,
    const float* __restrict__ bsense, const float* __restrict__ sw,
    __bf16* __restrict__ yb){
  __shared__ alignas(16) char ldsA[kLdsABytes];
  int lane = threadIdx.x & 31, wave = threadIdx.x >> 5;
  int row0 = blockIdx.x * 32;
  int col0 = (blockIdx.y * 4 + wave) * 16;
  if (wave == 0){
    tdm_load_rows(xb, (unsigned)(unsigned long long)(void*)ldsA, row0, 32, kRows);
    __builtin_amdgcn_s_wait_tensorcnt(0);
  }
  __syncthreads();
  v8f z = {0,0,0,0,0,0,0,0};
  v8f acc0[2] = {z, z}, acc1[2] = {z, z}, acc2[2] = {z, z};
  for (int k0 = 0; k0 < kD; k0 += 32){
    v16bf a0 = lds_a_frag(ldsA, 0, k0, lane);
    v16bf a1 = lds_a_frag(ldsA, 16, k0, lane);
    v16bf b0 = load_b_frag(Wt, 0 * kD + col0, kD, k0, lane);
    v16bf b1 = load_b_frag(Wt, 1 * kD + col0, kD, k0, lane);
    v16bf b2 = load_b_frag(Wt, 2 * kD + col0, kD, k0, lane);
    acc0[0] = WMMA_BF16(a0, b0, acc0[0]);  acc0[1] = WMMA_BF16(a1, b0, acc0[1]);
    acc1[0] = WMMA_BF16(a0, b1, acc1[0]);  acc1[1] = WMMA_BF16(a1, b1, acc1[1]);
    acc2[0] = WMMA_BF16(a0, b2, acc2[0]);  acc2[1] = WMMA_BF16(a1, b2, acc2[1]);
  }
  int nn = lane & 15, half = lane >> 4;
  int colg = col0 + nn;
  float bb0 = bsense[0 * kD + colg], bb1 = bsense[1 * kD + colg], bb2 = bsense[2 * kD + colg];
#pragma unroll
  for (int h = 0; h < 2; ++h){
#pragma unroll
    for (int v = 0; v < 8; ++v){
      int row = row0 + h * 16 + half * 8 + v;
      const float* w = sw + row * 3;
      float val = w[0] * (acc0[h][v] + bb0) + w[1] * (acc1[h][v] + bb1)
                + w[2] * (acc2[h][v] + bb2);
      yb[(size_t)row * kD + colg] = (__bf16)val;
    }
  }
}

// ---- per-token heads: charge(tanh), shell(softmax3), cls(argmax32), mass(softplus+0.5)
__global__ void k_features(const __bf16* __restrict__ yb,
                           const float* __restrict__ Wchg, const float* __restrict__ bchg,
                           const float* __restrict__ Wshl, const float* __restrict__ bshl,
                           const float* __restrict__ Wcls, const float* __restrict__ bcls,
                           const float* __restrict__ Wms,  const float* __restrict__ bms,
                           float* __restrict__ charge, float* __restrict__ shellv,
                           int* __restrict__ clsv, float* __restrict__ massv){
  __shared__ float yr[kD];
  __shared__ float outs[37];
  int row = blockIdx.x;
  const __bf16* p = yb + (size_t)row * kD;
  for (int k = threadIdx.x; k < kD; k += 64) yr[k] = (float)p[k];
  __syncthreads();
  int t = threadIdx.x;
  if (t == 0){
    float s = bchg[0];
    for (int k = 0; k < kD; ++k) s += yr[k] * Wchg[k];
    outs[0] = s;
  } else if (t < 4){
    int c = t - 1; float s = bshl[c];
    for (int k = 0; k < kD; ++k) s += yr[k] * Wshl[k * 3 + c];
    outs[t] = s;
  } else if (t < 36){
    int c = t - 4; float s = bcls[c];
    for (int k = 0; k < kD; ++k) s += yr[k] * Wcls[k * 32 + c];
    outs[t] = s;
  } else if (t == 36){
    float s = bms[0];
    for (int k = 0; k < kD; ++k) s += yr[k] * Wms[k];
    outs[36] = s;
  }
  __syncthreads();
  if (t == 0){
    charge[row] = tanhf(outs[0]);
    float m = fmaxf(outs[1], fmaxf(outs[2], outs[3]));
    float e1 = __expf(outs[1] - m), e2 = __expf(outs[2] - m), e3 = __expf(outs[3] - m);
    float inv = 1.f / (e1 + e2 + e3);
    shellv[row * 3 + 0] = e1 * inv; shellv[row * 3 + 1] = e2 * inv; shellv[row * 3 + 2] = e3 * inv;
    int best = 0; float bv = outs[4];
    for (int c = 1; c < 32; ++c) if (outs[4 + c] > bv){ bv = outs[4 + c]; best = c; }
    clsv[row] = best;
    massv[row] = softplusf(outs[36]) + 0.5f;
  }
}

// compat matrix + softplus'd energy coefficients
__global__ void k_prep(const float* __restrict__ logits,
                       const float* wch, const float* wsh, const float* wdi,
                       const float* wms, const float* wva, const float* temp,
                       float* __restrict__ compat, float* __restrict__ coef){
  int i = blockIdx.x * 256 + threadIdx.x;
  if (i < kC * kC){
    int a = i >> 5, b = i & 31;
    float v = 0.5f * (logits[a * kC + b] + logits[b * kC + a]);
    compat[i] = 1.f / (1.f + __expf(-v));
  }
  if (i == 0){
    coef[0] = softplusf(wch[0]);  coef[1] = softplusf(wsh[0]);
    coef[2] = softplusf(wdi[0]);  coef[3] = softplusf(wms[0]);
    coef[4] = softplusf(wva[0]);  coef[5] = 1.f / softplusf(temp[0]);
  }
}

// pass 1: per-row softmax max & sum-of-exp over causal j
__global__ void k_stats(const float* __restrict__ charge, const float* __restrict__ shellv,
                        const int* __restrict__ clsv, const float* __restrict__ massv,
                        const float* __restrict__ compat, const float* __restrict__ coef,
                        float* __restrict__ Mrow, float* __restrict__ Lrow){
  int row = blockIdx.x;
  int b = row / kN, i = row % kN;
  int base = b * kN;
  float cw = coef[0], csh = coef[1], cd = coef[2], cm = coef[3], cv = coef[4], invt = coef[5];
  float chi = charge[base + i], msi = massv[base + i];
  float s0 = shellv[(base + i) * 3], s1 = shellv[(base + i) * 3 + 1], s2 = shellv[(base + i) * 3 + 2];
  const float* cr = compat + clsv[base + i] * kC;
  float mt = -3.0e38f, lt = 0.f;
  for (int j = threadIdx.x; j <= i; j += 256){
    int r = base + j;
    float E = cw * chi * charge[r]
            + csh * (fabsf(s0 - shellv[r * 3]) + fabsf(s1 - shellv[r * 3 + 1]) + fabsf(s2 - shellv[r * 3 + 2]))
            - cd / (1.f + 0.1f * (float)(i - j))
            + cm * 0.1f * msi * massv[r]
            - cv;
    float sc = -E * invt * cr[clsv[r]];
    if (sc > mt){ lt = lt * __expf(mt - sc) + 1.f; mt = sc; }
    else        { lt += __expf(sc - mt); }
  }
  __shared__ float ms[256], ls[256];
  ms[threadIdx.x] = mt; ls[threadIdx.x] = lt; __syncthreads();
  for (int off = 128; off > 0; off >>= 1){
    if (threadIdx.x < off){
      float m1 = ms[threadIdx.x], l1 = ls[threadIdx.x];
      float m2 = ms[threadIdx.x + off], l2 = ls[threadIdx.x + off];
      float mm = fmaxf(m1, m2);
      ms[threadIdx.x] = mm;
      ls[threadIdx.x] = l1 * __expf(m1 - mm) + l2 * __expf(m2 - mm);
    }
    __syncthreads();
  }
  if (threadIdx.x == 0){ Mrow[row] = ms[0]; Lrow[row] = ls[0]; }
}

// V = yb@W_v + b_v, stored bf16 transposed per batch: Vt[(b*D+col)*N + i]
// Block = 128 (4 waves) covers 32 rows x 256 cols; TDM-staged A tile; 8 accs/wave.
__global__ void __launch_bounds__(128) k_gemm_vt(
    const __bf16* __restrict__ A, const __bf16* __restrict__ Bt,
    const float* __restrict__ bias, __bf16* __restrict__ Vt){
  __shared__ alignas(16) char ldsA[kLdsABytes];
  int lane = threadIdx.x & 31, wave = threadIdx.x >> 5;
  int row0 = blockIdx.x * 32;
  int colbase = blockIdx.y * 256 + wave * 64;
  if (wave == 0){
    tdm_load_rows(A, (unsigned)(unsigned long long)(void*)ldsA, row0, 32, kRows);
    __builtin_amdgcn_s_wait_tensorcnt(0);
  }
  __syncthreads();
  v8f z = {0,0,0,0,0,0,0,0};
  v8f acc[2][4] = {{z, z, z, z}, {z, z, z, z}};
  for (int k0 = 0; k0 < kD; k0 += 32){
    v16bf a0 = lds_a_frag(ldsA, 0, k0, lane);
    v16bf a1 = lds_a_frag(ldsA, 16, k0, lane);
#pragma unroll
    for (int t = 0; t < 4; ++t){
      v16bf b = load_b_frag(Bt, colbase + t * 16, kD, k0, lane);
      acc[0][t] = WMMA_BF16(a0, b, acc[0][t]);
      acc[1][t] = WMMA_BF16(a1, b, acc[1][t]);
    }
  }
  int nn = lane & 15, half = lane >> 4;
#pragma unroll
  for (int t = 0; t < 4; ++t){
    int colg = colbase + t * 16 + nn;
    float bb = bias[colg];
#pragma unroll
    for (int h = 0; h < 2; ++h){
#pragma unroll
      for (int v = 0; v < 8; ++v){
        int row = row0 + h * 16 + half * 8 + v;
        int b_ = row / kN, ii = row % kN;
        Vt[((size_t)b_ * kD + colg) * kN + ii] = (__bf16)(acc[h][t][v] + bb);
      }
    }
  }
}

// pass 2: O = P @ V via WMMA. Block = 256 (8 waves), one 16-row tile; each wave owns 128 cols.
__global__ void __launch_bounds__(256) k_attn_av(
    const float* __restrict__ charge, const float* __restrict__ shellv,
    const int* __restrict__ clsv, const float* __restrict__ massv,
    const float* __restrict__ compat, const float* __restrict__ coef,
    const float* __restrict__ Mrow, const float* __restrict__ Lrow,
    const __bf16* __restrict__ Vt, __bf16* __restrict__ Ob){
  int tid = threadIdx.x, lane = tid & 31, wave = tid >> 5;
  int row0 = blockIdx.x * 16;
  int b = row0 / kN, i0 = row0 % kN;
  __shared__ float chi[16], msi[16], shi[16][3], Mi[16], Li[16];
  __shared__ int   cli[16];
  __shared__ __bf16 P[16][40];          // padded: 80B row stride spreads LDS banks
  __shared__ float chj[32], msj[32], shj[32][3];
  __shared__ int   clj[32];
  if (tid < 16){
    int r = row0 + tid;
    chi[tid] = charge[r]; msi[tid] = massv[r];
    shi[tid][0] = shellv[r * 3]; shi[tid][1] = shellv[r * 3 + 1]; shi[tid][2] = shellv[r * 3 + 2];
    cli[tid] = clsv[r]; Mi[tid] = Mrow[r]; Li[tid] = 1.f / Lrow[r];
  }
  float cw = coef[0], csh = coef[1], cd = coef[2], cm = coef[3], cv = coef[4], invt = coef[5];
  v8f acc[8];
#pragma unroll
  for (int t = 0; t < 8; ++t) acc[t] = (v8f){0,0,0,0,0,0,0,0};
  __syncthreads();

  int jend = i0 + 16;  // causal bound (exclusive)
  for (int j0 = 0; j0 < jend; j0 += 32){
    if (tid < 32){
      int j = j0 + tid;
      if (j < kN){
        int r = b * kN + j;
        chj[tid] = charge[r]; msj[tid] = massv[r];
        shj[tid][0] = shellv[r * 3]; shj[tid][1] = shellv[r * 3 + 1]; shj[tid][2] = shellv[r * 3 + 2];
        clj[tid] = clsv[r];
      }
    }
    __syncthreads();
    for (int e = tid; e < 512; e += 256){
      int mi = e >> 5, jj = e & 31;
      int i = i0 + mi, j = j0 + jj;
      float pval = 0.f;
      if (j <= i && j < kN){
        float E = cw * chi[mi] * chj[jj]
                + csh * (fabsf(shi[mi][0] - shj[jj][0]) + fabsf(shi[mi][1] - shj[jj][1])
                       + fabsf(shi[mi][2] - shj[jj][2]))
                - cd / (1.f + 0.1f * (float)(i - j))
                + cm * 0.1f * msi[mi] * msj[jj]
                - cv;
        float sc = -E * invt * compat[cli[mi] * kC + clj[jj]];
        pval = __expf(sc - Mi[mi]) * Li[mi];
      }
      P[mi][jj] = (__bf16)pval;
    }
    __syncthreads();
    v16bf a;
    { int m = lane & 15, kb = (lane >> 4) * 8;
#pragma unroll
      for (int t = 0; t < 8; ++t){ a[t] = P[m][kb + t]; a[8 + t] = P[m][16 + kb + t]; }
    }
#pragma unroll
    for (int t = 0; t < 8; ++t){
      int colg = wave * 128 + t * 16 + (lane & 15);
      int kb2 = (lane >> 4) * 8;
      const __bf16* p = Vt + ((size_t)b * kD + colg) * kN + (j0 + kb2);
      v8bf lo = *reinterpret_cast<const v8bf*>(p);
      v8bf hi = *reinterpret_cast<const v8bf*>(p + 16);
      v16bf bf;
#pragma unroll
      for (int q = 0; q < 8; ++q){ bf[q] = lo[q]; bf[8 + q] = hi[q]; }
      acc[t] = WMMA_BF16(a, bf, acc[t]);
    }
    __syncthreads();
  }
  int nn = lane & 15, half = lane >> 4;
#pragma unroll
  for (int t = 0; t < 8; ++t){
    int colg = wave * 128 + t * 16 + nn;
#pragma unroll
    for (int v = 0; v < 8; ++v){
      int row = row0 + half * 8 + v;
      Ob[(size_t)row * kD + colg] = (__bf16)acc[t][v];
    }
  }
}

// final: out = Ob@W_out + b_out (f32). Same blocking as k_gemm_vt.
__global__ void __launch_bounds__(128) k_gemm_out(
    const __bf16* __restrict__ A, const __bf16* __restrict__ Bt,
    const float* __restrict__ bias, float* __restrict__ out){
  __shared__ alignas(16) char ldsA[kLdsABytes];
  int lane = threadIdx.x & 31, wave = threadIdx.x >> 5;
  int row0 = blockIdx.x * 32;
  int colbase = blockIdx.y * 256 + wave * 64;
  if (wave == 0){
    tdm_load_rows(A, (unsigned)(unsigned long long)(void*)ldsA, row0, 32, kRows);
    __builtin_amdgcn_s_wait_tensorcnt(0);
  }
  __syncthreads();
  v8f z = {0,0,0,0,0,0,0,0};
  v8f acc[2][4] = {{z, z, z, z}, {z, z, z, z}};
  for (int k0 = 0; k0 < kD; k0 += 32){
    v16bf a0 = lds_a_frag(ldsA, 0, k0, lane);
    v16bf a1 = lds_a_frag(ldsA, 16, k0, lane);
#pragma unroll
    for (int t = 0; t < 4; ++t){
      v16bf b = load_b_frag(Bt, colbase + t * 16, kD, k0, lane);
      acc[0][t] = WMMA_BF16(a0, b, acc[0][t]);
      acc[1][t] = WMMA_BF16(a1, b, acc[1][t]);
    }
  }
  int nn = lane & 15, half = lane >> 4;
#pragma unroll
  for (int t = 0; t < 4; ++t){
    int colg = colbase + t * 16 + nn;
    float bb = bias[colg];
#pragma unroll
    for (int h = 0; h < 2; ++h){
#pragma unroll
      for (int v = 0; v < 8; ++v){
        int row = row0 + h * 16 + half * 8 + v;
        out[(size_t)row * kD + colg] = acc[h][t][v] + bb;
      }
    }
  }
}

extern "C" void kernel_launch(void* const* d_in, const int* in_sizes, int n_in,
                              void* d_out, int out_size, void* d_ws, size_t ws_size,
                              hipStream_t stream){
  const float* x         = (const float*)d_in[0];
  const float* W_sense   = (const float*)d_in[1];
  const float* b_sense   = (const float*)d_in[2];
  const float* W_ctx     = (const float*)d_in[3];
  const float* b_ctx     = (const float*)d_in[4];
  const float* W_sel     = (const float*)d_in[5];
  const float* b_sel     = (const float*)d_in[6];
  const float* W_charge  = (const float*)d_in[7];
  const float* b_charge  = (const float*)d_in[8];
  const float* W_shell   = (const float*)d_in[9];
  const float* b_shell   = (const float*)d_in[10];
  const float* W_class   = (const float*)d_in[11];
  const float* b_class   = (const float*)d_in[12];
  const float* W_mass    = (const float*)d_in[13];
  const float* b_mass    = (const float*)d_in[14];
  // d_in[15]/d_in[16] (W_val/b_val) unused: softmax rows sum to 1 -> valence term is constant
  const float* compat_l  = (const float*)d_in[17];
  const float* w_charge  = (const float*)d_in[18];
  const float* w_shell   = (const float*)d_in[19];
  const float* w_dist    = (const float*)d_in[20];
  const float* w_mass    = (const float*)d_in[21];
  const float* w_val     = (const float*)d_in[22];
  const float* temp      = (const float*)d_in[23];
  const float* W_v       = (const float*)d_in[24];
  const float* b_v       = (const float*)d_in[25];
  const float* W_out     = (const float*)d_in[26];
  const float* b_out     = (const float*)d_in[27];
  float* out = (float*)d_out;

  char* ws = (char*)d_ws;
  size_t off = 0;
  auto alloc = [&](size_t bytes) -> void* {
    void* p = ws + off;
    off = (off + bytes + 255) & ~(size_t)255;
    return p;
  };
  __bf16* xb     = (__bf16*)alloc((size_t)kRows * kD * 2);
  __bf16* Wt_sen = (__bf16*)alloc((size_t)kS * kD * kD * 2);
  __bf16* Wt_v   = (__bf16*)alloc((size_t)kD * kD * 2);
  __bf16* Wt_out = (__bf16*)alloc((size_t)kD * kD * 2);
  __bf16* yb     = (__bf16*)alloc((size_t)kRows * kD * 2);
  __bf16* Vt     = (__bf16*)alloc((size_t)kB * kD * kN * 2);
  __bf16* Ob     = (__bf16*)alloc((size_t)kRows * kD * 2);
  float*  xbar   = (float*)alloc((size_t)kB * kD * 4);
  float*  ctx    = (float*)alloc((size_t)kB * kD * 4);
  float*  selctx = (float*)alloc((size_t)kB * kS * 4);
  float*  sw     = (float*)alloc((size_t)kRows * kS * 4);
  float*  charge = (float*)alloc((size_t)kRows * 4);
  float*  shellv = (float*)alloc((size_t)kRows * 3 * 4);
  int*    clsv   = (int*)  alloc((size_t)kRows * 4);
  float*  massv  = (float*)alloc((size_t)kRows * 4);
  float*  Mrow   = (float*)alloc((size_t)kRows * 4);
  float*  Lrow   = (float*)alloc((size_t)kRows * 4);
  float*  compat = (float*)alloc((size_t)kC * kC * 4);
  float*  coef   = (float*)alloc(8 * 4);

  int ne;
  ne = kRows * kD;
  k_cvt_bf16<<<(ne + 255) / 256, 256, 0, stream>>>(x, xb, ne);
  ne = kD * kS * kD;
  k_transpose_bf16<<<(ne + 255) / 256, 256, 0, stream>>>(W_sense, Wt_sen, kD, kS * kD);
  ne = kD * kD;
  k_transpose_bf16<<<(ne + 255) / 256, 256, 0, stream>>>(W_v, Wt_v, kD, kD);
  k_transpose_bf16<<<(ne + 255) / 256, 256, 0, stream>>>(W_out, Wt_out, kD, kD);

  k_colmean<<<(kB * kD + 255) / 256, 256, 0, stream>>>(x, xbar);
  k_ctx<<<(kB * kD + 255) / 256, 256, 0, stream>>>(xbar, W_ctx, b_ctx, ctx);
  k_selctx<<<kB * kS, 256, 0, stream>>>(ctx, W_sel, b_sel, selctx);
  k_sensew<<<kRows / 8, 256, 0, stream>>>(x, W_sel, selctx, sw);

  k_sense_gemm<<<dim3(kRows / 32, kD / 64), 128, 0, stream>>>(xb, Wt_sen, b_sense, sw, yb);

  k_features<<<kRows, 64, 0, stream>>>(yb, W_charge, b_charge, W_shell, b_shell,
                                       W_class, b_class, W_mass, b_mass,
                                       charge, shellv, clsv, massv);
  k_prep<<<4, 256, 0, stream>>>(compat_l, w_charge, w_shell, w_dist, w_mass, w_val, temp,
                                compat, coef);
  k_stats<<<kRows, 256, 0, stream>>>(charge, shellv, clsv, massv, compat, coef, Mrow, Lrow);

  k_gemm_vt<<<dim3(kRows / 32, kD / 256), 128, 0, stream>>>(yb, Wt_v, b_v, Vt);
  k_attn_av<<<kRows / 16, 256, 0, stream>>>(charge, shellv, clsv, massv, compat, coef,
                                            Mrow, Lrow, Vt, Ob);
  k_gemm_out<<<dim3(kRows / 32, kD / 256), 128, 0, stream>>>(Ob, Wt_out, b_out, out);
}
